// TransformerTransferGSCPredictor_35218731827321
// MI455X (gfx1250) — matrix operand order repaired and checked by reading, not compiled, not run on hardware
//
#include <hip/hip_runtime.h>
#include <math.h>

// ---------------------------------------------------------------------------
// Types for CDNA5 WMMA
// ---------------------------------------------------------------------------
typedef __attribute__((ext_vector_type(16))) __bf16 v16bf;
typedef __attribute__((ext_vector_type(8)))  float  v8f;

union FragBF {
    v16bf v;
    uint4 q[2];
};

__device__ __forceinline__ unsigned short f2bf(float f) {
    unsigned u = __builtin_bit_cast(unsigned, f);
    u += 0x7FFFu + ((u >> 16) & 1u);          // round-to-nearest-even
    return (unsigned short)(u >> 16);
}

// ---------------------------------------------------------------------------
// Staging helpers (templated on tile ROWS = 64 or 128, K-chunk fixed at 64).
// A ROWSx64 fp32 tile is loaded to registers then stored to LDS as bf16 in
// [row][k] layout (row-padded to 72 ushorts for bank spread / alignment).
//
//  direct: source row-major [rows, K]  -> lds[row][k]     (A normal, B [N,K])
//  trans : source row-major [K, cols]  -> lds[col][k]     (A^T,      B [K,N])
//
// Fast path = whole tile in-bounds and 16B-alignable (ld % 4 == 0):
//   direct: global_load_b128 + ds_store_b64 (2xbf16 packed per dword)
//   trans : paired-row global_load_b128 + ds_store_b32 (k-pair packed)
// ---------------------------------------------------------------------------
template <int ROWS>
__device__ __forceinline__ void ld_direct(float4* r, const float* __restrict__ src,
                                          int ld, int rowBase, int rowBound,
                                          int k0, int K, int tid, bool vec)
{
    if (vec && (rowBase + ROWS <= rowBound) && (k0 + 64 <= K)) {
#pragma unroll
        for (int it = 0; it < ROWS / 16; ++it) {
            int e = tid + it * 256, rr = e >> 4, c4 = (e & 15) * 4;
            r[it] = *(const float4*)&src[(size_t)(rowBase + rr) * ld + k0 + c4];
        }
    } else {
#pragma unroll
        for (int it = 0; it < ROWS / 16; ++it) {
            int e = tid + it * 256, rr = e >> 4, c4 = (e & 15) * 4;
            int gr = rowBase + rr, gk = k0 + c4;
            float4 v = {0.f, 0.f, 0.f, 0.f};
            float* vp = &v.x;
            if (gr < rowBound) {
#pragma unroll
                for (int j = 0; j < 4; ++j)
                    if (gk + j < K) vp[j] = src[(size_t)gr * ld + gk + j];
            }
            r[it] = v;
        }
    }
}

template <int ROWS>
__device__ __forceinline__ void st_direct(unsigned short (*dst)[72],
                                          const float4* r, int tid)
{
#pragma unroll
    for (int it = 0; it < ROWS / 16; ++it) {
        int e = tid + it * 256, rr = e >> 4, c4 = (e & 15) * 4;
        uint2 u;
        u.x = (unsigned)f2bf(r[it].x) | ((unsigned)f2bf(r[it].y) << 16);
        u.y = (unsigned)f2bf(r[it].z) | ((unsigned)f2bf(r[it].w) << 16);
        *(uint2*)&dst[rr][c4] = u;
    }
}

template <int ROWS>
__device__ __forceinline__ void ld_trans(float4* r, const float* __restrict__ src,
                                         int ld, int k0, int K,
                                         int colBase, int colBound, int tid, bool vec)
{
    constexpr int CG = ROWS / 4;               // float4 column groups per k-row
    if (vec && (colBase + ROWS <= colBound) && (k0 + 64 <= K)) {
#pragma unroll
        for (int it = 0; it < ROWS / 32; ++it) {
            int e = tid + it * 256, k2 = e / CG, c4 = (e % CG) * 4;
            int gk = k0 + 2 * k2;
            r[2 * it + 0] = *(const float4*)&src[(size_t)gk * ld + colBase + c4];
            r[2 * it + 1] = *(const float4*)&src[(size_t)(gk + 1) * ld + colBase + c4];
        }
    } else {
#pragma unroll
        for (int it = 0; it < ROWS / 32; ++it) {
            int e = tid + it * 256, k2 = e / CG, c4 = (e % CG) * 4;
            int gk0 = k0 + 2 * k2, gk1 = gk0 + 1;
            float4 v0 = {0.f, 0.f, 0.f, 0.f}, v1 = v0;
            float* p0 = &v0.x; float* p1 = &v1.x;
#pragma unroll
            for (int j = 0; j < 4; ++j) {
                int col = colBase + c4 + j;
                if (col < colBound) {
                    if (gk0 < K) p0[j] = src[(size_t)gk0 * ld + col];
                    if (gk1 < K) p1[j] = src[(size_t)gk1 * ld + col];
                }
            }
            r[2 * it + 0] = v0;
            r[2 * it + 1] = v1;
        }
    }
}

template <int ROWS>
__device__ __forceinline__ void st_trans(unsigned short (*dst)[72],
                                         const float4* r, int tid)
{
    constexpr int CG = ROWS / 4;
#pragma unroll
    for (int it = 0; it < ROWS / 32; ++it) {
        int e = tid + it * 256, k2 = e / CG, c4 = (e % CG) * 4;
        const float* lo = &r[2 * it + 0].x;
        const float* hi = &r[2 * it + 1].x;
#pragma unroll
        for (int j = 0; j < 4; ++j) {
            unsigned u = (unsigned)f2bf(lo[j]) | ((unsigned)f2bf(hi[j]) << 16);
            *(unsigned*)&dst[c4 + j][2 * k2] = u;
        }
    }
}

// ---------------------------------------------------------------------------
// Tiled GEMM:  C[M,N] = act( op(A) @ op(B) + bias + addsrc )
//   a_kt==0: A row-major [M,K] (lda = row stride)
//   a_kt==1: A row-major [K,M] (computes A^T @ B; lda = row stride of source)
//   b_nk==0: B row-major [K,N] (ldb = N);  b_nk==1: B row-major [N,K] (ldb = K)
// 128x64x64 block tile, 8 waves in a 4(m)x2(n) grid, 2x2 accumulator tiles
// per wave, double-buffered LDS (one barrier per K chunk), bf16 operands /
// fp32 accumulate via v_wmma_f32_16x16x32_bf16 (8 WMMAs per chunk per wave).
// ---------------------------------------------------------------------------
__launch_bounds__(256)
__global__ void tg_gemm_wmma_bf16(const float* __restrict__ A, int lda, int a_kt,
                                  const float* __restrict__ B, int ldb, int b_nk,
                                  const float* __restrict__ bias,
                                  const float* __restrict__ addsrc, int ldadd,
                                  float* __restrict__ C, int ldc,
                                  int M, int N, int K, int act)
{
    __shared__ __attribute__((aligned(16))) unsigned short lA[2][128][72]; // [m][k]
    __shared__ __attribute__((aligned(16))) unsigned short lB[2][64][72];  // [n][k]

    const int tid  = threadIdx.x;
    const int lane = tid & 31;
    const int wid  = tid >> 5;
    const int hl   = lane >> 4;     // K-half selector (lanes 0-15 vs 16-31)
    const int l16  = lane & 15;

    const int bm = blockIdx.y * 128;
    const int bn = blockIdx.x * 64;
    const int wm = (wid & 3) * 32;           // wave's 32-row slab within tile
    const int wn = (wid >> 2) * 32;          // wave's 32-col slab

    const bool aVec = ((lda & 3) == 0);
    const bool bVec = ((ldb & 3) == 0);

    v8f acc00 = {0.f,0.f,0.f,0.f,0.f,0.f,0.f,0.f};
    v8f acc01 = acc00, acc10 = acc00, acc11 = acc00;

    float4 ar[8], br[4];
    const int nch = (K + 63) >> 6;

    auto loadStage = [&](int k0) {
        if (!a_kt) ld_direct<128>(ar, A, lda, bm, M, k0, K, tid, aVec);
        else       ld_trans <128>(ar, A, lda, k0, K, bm, M, tid, aVec);
        if (!b_nk) ld_trans <64>(br, B, ldb, k0, K, bn, N, tid, bVec);
        else       ld_direct<64>(br, B, ldb, bn, N, k0, K, tid, bVec);
    };
    auto storeStage = [&](int buf) {
        if (!a_kt) st_direct<128>(lA[buf], ar, tid); else st_trans<128>(lA[buf], ar, tid);
        if (!b_nk) st_trans <64>(lB[buf], br, tid); else st_direct<64>(lB[buf], br, tid);
    };

    loadStage(0);
    storeStage(0);
    __syncthreads();

    for (int i = 0; i < nch; ++i) {
        const int cur = i & 1;
        const bool more = (i + 1 < nch);
        if (more) loadStage((i + 1) << 6);    // global loads overlap compute

#pragma unroll
        for (int kk = 0; kk < 64; kk += 32) {
            FragBF a0, a1, b0, b1;
            // 16-bit A 16x32 layout: lanes0-15 K {kk..+7, kk+16..+23},
            //                        lanes16-31 K {kk+8..+15, kk+24..+31}
            a0.q[0] = *(const uint4*)&lA[cur][wm + l16][kk + hl * 8];
            a0.q[1] = *(const uint4*)&lA[cur][wm + l16][kk + 16 + hl * 8];
            a1.q[0] = *(const uint4*)&lA[cur][wm + 16 + l16][kk + hl * 8];
            a1.q[1] = *(const uint4*)&lA[cur][wm + 16 + l16][kk + 16 + hl * 8];
            // 16-bit B 32x16 layout: lanes0-15 K {kk..+7, kk+16..+23},
            //                        lanes16-31 K {kk+8..+15, kk+24..+31}
            b0.q[0] = *(const uint4*)&lB[cur][wn + l16][kk + hl * 8];
            b0.q[1] = *(const uint4*)&lB[cur][wn + l16][kk + 16 + hl * 8];
            b1.q[0] = *(const uint4*)&lB[cur][wn + 16 + l16][kk + hl * 8];
            b1.q[1] = *(const uint4*)&lB[cur][wn + 16 + l16][kk + 16 + hl * 8];

            acc00 = __builtin_amdgcn_wmma_f32_16x16x32_bf16(
                        false, a0.v, false, b0.v, (short)0, acc00, false, false);
            acc01 = __builtin_amdgcn_wmma_f32_16x16x32_bf16(
                        false, a0.v, false, b1.v, (short)0, acc01, false, false);
            acc10 = __builtin_amdgcn_wmma_f32_16x16x32_bf16(
                        false, a1.v, false, b0.v, (short)0, acc10, false, false);
            acc11 = __builtin_amdgcn_wmma_f32_16x16x32_bf16(
                        false, a1.v, false, b1.v, (short)0, acc11, false, false);
        }

        if (more) {
            storeStage(1 - cur);
            __syncthreads();
        }
    }

    // ---- epilogue: C/D layout: VGPR r -> M = r + 8*hl, N = l16 ----
    auto emit = [&](const v8f& acc, int mBase, int nBase) {
#pragma unroll
        for (int r = 0; r < 8; ++r) {
            int m = mBase + r + hl * 8;
            int n = nBase + l16;
            if (m < M && n < N) {
                float v = acc[r];
                if (bias)   v += bias[n];
                if (addsrc) v += addsrc[(size_t)m * ldadd + n];
                if (act)    v = fmaxf(v, 0.f);
                C[(size_t)m * ldc + n] = v;
            }
        }
    };
    emit(acc00, bm + wm,      bn + wn);
    emit(acc01, bm + wm,      bn + wn + 16);
    emit(acc10, bm + wm + 16, bn + wn);
    emit(acc11, bm + wm + 16, bn + wn + 16);
}

// ---------------------------------------------------------------------------
// Row softmax (numerically stable), one block of 128 threads per row.
// ---------------------------------------------------------------------------
__global__ void tg_softmax_rows(const float* __restrict__ in,
                                float* __restrict__ out, int cols)
{
    int row = blockIdx.x, t = threadIdx.x;
    const float* r = in + (size_t)row * cols;
    float* o = out + (size_t)row * cols;
    __shared__ float red[128];

    float mx = -3.4e38f;
    for (int c = t; c < cols; c += 128) mx = fmaxf(mx, r[c]);
    red[t] = mx; __syncthreads();
    for (int s = 64; s > 0; s >>= 1) { if (t < s) red[t] = fmaxf(red[t], red[t + s]); __syncthreads(); }
    mx = red[0]; __syncthreads();

    float sum = 0.f;
    for (int c = t; c < cols; c += 128) sum += expf(r[c] - mx);
    red[t] = sum; __syncthreads();
    for (int s = 64; s > 0; s >>= 1) { if (t < s) red[t] += red[t + s]; __syncthreads(); }
    float inv = 1.f / red[0];

    for (int c = t; c < cols; c += 128) o[c] = expf(r[c] - mx) * inv;
}

__global__ void tg_zero(float* __restrict__ p, size_t n)
{
    for (size_t i = (size_t)blockIdx.x * blockDim.x + threadIdx.x; i < n;
         i += (size_t)gridDim.x * blockDim.x) p[i] = 0.f;
}

// segment_sum(ms_ip, subj) / (.., obj): scatter-add into cat cols [0,256)/[256,512)
__global__ void tg_scatter_pairs(const int* __restrict__ pairs,
                                 const float* __restrict__ msip,
                                 float* __restrict__ cat, int ldc)
{
    int p = blockIdx.x, c = threadIdx.x;
    int s = pairs[2 * p], o = pairs[2 * p + 1];
    float v = msip[(size_t)p * 256 + c];
    atomicAdd(&cat[(size_t)s * ldc + c], v);
    atomicAdd(&cat[(size_t)o * ldc + 256 + c], v);
}

// cat[p, 0:256) = ms_ie[subj[p]], cat[p, 256:512) = ms_ie[obj[p]]
__global__ void tg_gather_pairs(const int* __restrict__ pairs,
                                const float* __restrict__ msie,
                                float* __restrict__ cat, int ldc)
{
    int p = blockIdx.x, c = threadIdx.x;
    int s = pairs[2 * p], o = pairs[2 * p + 1];
    cat[(size_t)p * ldc + c]       = msie[(size_t)s * 256 + c];
    cat[(size_t)p * ldc + 256 + c] = msie[(size_t)o * 256 + c];
}

// rx = sigmoid(t_r) * x   (safe in-place over t_r)
__global__ void tg_gru_rx(const float* __restrict__ tr, const float* __restrict__ x,
                          float* __restrict__ rx, size_t n)
{
    size_t i = (size_t)blockIdx.x * 256 + threadIdx.x;
    if (i < n) {
        float r = 1.f / (1.f + expf(-tr[i]));
        rx[i] = r * x[i];
    }
}

// z = sigmoid(t_z);  x = (1-z)*x + z*tanh(t_h)   (in-place x)
__global__ void tg_gru_out(const float* __restrict__ tz, const float* __restrict__ th,
                           float* __restrict__ x, size_t n)
{
    size_t i = (size_t)blockIdx.x * 256 + threadIdx.x;
    if (i < n) {
        float z = 1.f / (1.f + expf(-tz[i]));
        x[i] = (1.f - z) * x[i] + z * tanhf(th[i]);
    }
}

// ---------------------------------------------------------------------------
// Host orchestration
// ---------------------------------------------------------------------------
static const int H = 1024, H4 = 256, CE = 151, CP = 51, EMB = 300;
static const int NE = 4096, NPRED = 8192, T = 3, STEPS = 3, FEAT = 4096, RW = 1792;

extern "C" void kernel_launch(void* const* d_in, const int* in_sizes, int n_in,
                              void* d_out, int out_size, void* d_ws, size_t ws_size,
                              hipStream_t stream)
{
    (void)in_sizes; (void)n_in; (void)out_size; (void)ws_size;

    const float* roi   = (const float*)d_in[0];
    const float* uni   = (const float*)d_in[1];
    const float* objl  = (const float*)d_in[2];
    const float* a_e2e = (const float*)d_in[3];
    const float* a_e2p = (const float*)d_in[4];
    const float* a_p2e = (const float*)d_in[5];
    const float* a_p2p = (const float*)d_in[6];
    const float* emb_e = (const float*)d_in[7];
    const float* emb_p = (const float*)d_in[8];

    // params flattened jax-tree style (dict keys alphabetical; {'b','w'} -> b,w;
    // gru {u3,u4,u5,w3,w4,w5} each {b,w}; mlp2 list -> [l0.b,l0.w,l1.b,l1.w])
    const int P0 = 9;
    const int GRU_IMG_ENT = P0 + 0, GRU_IMG_PRED = P0 + 12;
    const int GRU_ONT_ENT = P0 + 24, GRU_ONT_PRED = P0 + 36;
    const int L_INIT_OE = P0 + 48, L_INIT_OP = P0 + 50, L_OBJ = P0 + 52;
    const int M_OUT_IP = P0 + 54, M_OUT_OP = P0 + 58;
    const int M_RECV_IE = P0 + 62, M_RECV_IP = P0 + 66;
    const int M_RECV_OE = P0 + 70, M_RECV_OP = P0 + 74;
    const int L_REL = P0 + 78;
    const int M_SEND_IE = P0 + 80, M_SEND_IP = P0 + 84;
    const int M_SEND_OE = P0 + 88, M_SEND_OP = P0 + 92;
    const int* pairs = (const int*)d_in[P0 + 96];

    auto F = [&](int i) { return (const float*)d_in[i]; };
    float* out = (float*)d_out;

    // ---- workspace bump allocator ----
    float* ws = (float*)d_ws;
    size_t off = 0;
    auto alloc = [&](size_t n) { float* p = ws + off; off += (n + 63) & ~(size_t)63; return p; };

    float* ont_ent  = alloc((size_t)CE * H);
    float* ont_pred = alloc((size_t)CP * H);
    float* img_ent  = alloc((size_t)NE * H);
    float* img_pred = alloc((size_t)NPRED * H);
    float* i2o_ent  = alloc((size_t)NE * CE);
    float* i2o_pred = alloc((size_t)NPRED * CP);
    float* ms_oe    = alloc((size_t)CE * H4);
    float* ms_op    = alloc((size_t)CP * H4);
    float* ms_ie    = alloc((size_t)NE * H4);
    float* ms_ip    = alloc((size_t)NPRED * H4);
    float* cat_o    = alloc((size_t)CE * RW);   // ont cat / out-ont hidden
    float* hid_o    = alloc((size_t)CE * RW);   // ont hidden / Oop
    float* mr_oe    = alloc((size_t)CE * H);
    float* mr_op    = alloc((size_t)CP * H);
    float* SA = alloc((size_t)NPRED * H);       // hiddens / gru t_z
    float* SB = alloc((size_t)NPRED * H);       // cat_img / gru t_r / Oip
    float* SC = alloc((size_t)NPRED * H);       // mr_ie
    float* SD = alloc((size_t)NPRED * H);       // mr_ip
    float* SE = alloc((size_t)NPRED * H);       // gru t_h

    // ---- launch helpers ----
    auto gemm = [&](const float* A, int lda, int a_kt, int M, int K,
                    const float* B, int ldb, int b_nk, int N,
                    const float* bias, const float* addsrc, int ldadd,
                    float* C, int ldc, int act) {
        dim3 g((N + 63) / 64, (M + 127) / 128), b(256);
        tg_gemm_wmma_bf16<<<g, b, 0, stream>>>(A, lda, a_kt, B, ldb, b_nk, bias,
                                               addsrc, ldadd, C, ldc, M, N, K, act);
    };
    auto mlp2 = [&](const float* X, int M, int Din, int Dh, int Dout, int p,
                    float* hid, float* o, int ldo, bool lastRelu) {
        gemm(X, Din, 0, M, Din, F(p + 1), Dh, 0, Dh, F(p + 0), nullptr, 0, hid, Dh, 1);
        gemm(hid, Dh, 0, M, Dh, F(p + 3), Dout, 0, Dout, F(p + 2), nullptr, 0, o, ldo,
             lastRelu ? 1 : 0);
    };
    // C[M,256] = src^T @ msg  (src row-major [K, M], msg row-major [K, 256])
    auto atg = [&](const float* src, int lds_, int M, int K, const float* msg,
                   float* C, int ldc) {
        gemm(src, lds_, 1, M, K, msg, H4, 0, H4, nullptr, nullptr, 0, C, ldc, 0);
    };
    auto zero = [&](float* p, size_t n) {
        tg_zero<<<dim3((unsigned)((n + 255) / 256)), dim3(256), 0, stream>>>(p, n);
    };
    auto gru = [&](int g, float* x, const float* m, int M,
                   float* tz, float* tr, float* th) {
        // leaves: u3b,u3w,u4b,u4w,u5b,u5w,w3b,w3w,w4b,w4w,w5b,w5w
        gemm(m, H, 0, M, H, F(g + 7), H, 0, H, F(g + 6), nullptr, 0, tz, H, 0);
        gemm(x, H, 0, M, H, F(g + 1), H, 0, H, F(g + 0), tz, H,   tz, H, 0);
        gemm(m, H, 0, M, H, F(g + 9), H, 0, H, F(g + 8), nullptr, 0, tr, H, 0);
        gemm(x, H, 0, M, H, F(g + 3), H, 0, H, F(g + 2), tr, H,   tr, H, 0);
        gemm(m, H, 0, M, H, F(g + 11), H, 0, H, F(g + 10), nullptr, 0, th, H, 0);
        size_t n = (size_t)M * H;
        tg_gru_rx<<<dim3((unsigned)((n + 255) / 256)), dim3(256), 0, stream>>>(tr, x, tr, n);
        gemm(tr, H, 0, M, H, F(g + 5), H, 0, H, F(g + 4), th, H, th, H, 0);
        tg_gru_out<<<dim3((unsigned)((n + 255) / 256)), dim3(256), 0, stream>>>(tz, th, x, n);
    };

    // ---- initial projections ----
    gemm(emb_e, EMB, 0, CE, EMB, F(L_INIT_OE + 1), H, 0, H, F(L_INIT_OE), nullptr, 0, ont_ent, H, 0);
    gemm(emb_p, EMB, 0, CP, EMB, F(L_INIT_OP + 1), H, 0, H, F(L_INIT_OP), nullptr, 0, ont_pred, H, 0);
    gemm(roi, FEAT, 0, NE, FEAT, F(L_OBJ + 1), H, 0, H, F(L_OBJ), nullptr, 0, img_ent, H, 0);
    gemm(uni, FEAT, 0, NPRED, FEAT, F(L_REL + 1), H, 0, H, F(L_REL), nullptr, 0, img_pred, H, 0);
    tg_softmax_rows<<<dim3(NE), dim3(128), 0, stream>>>(objl, i2o_ent, CE);
    zero(i2o_pred, (size_t)NPRED * CP);

    for (int step = 0; step < STEPS; ++step) {
        // ---- send MLPs ----
        mlp2(ont_ent,  CE,    H, H / 2, H4, M_SEND_OE, SA, ms_oe, H4, true);
        mlp2(ont_pred, CP,    H, H / 2, H4, M_SEND_OP, SA, ms_op, H4, true);
        mlp2(img_ent,  NE,    H, H / 2, H4, M_SEND_IE, SA, ms_ie, H4, true);
        mlp2(img_pred, NPRED, H, H / 2, H4, M_SEND_IP, SA, ms_ip, H4, true);

        // ---- recv ont ent: cat = [agg(e2e,ms_oe) | agg(p2e,ms_op) | i2o_ent^T@ms_ie]
        for (int t = 0; t < T; ++t)
            atg(a_e2e + (size_t)t * CE * CE, CE, CE, CE, ms_oe, cat_o + t * H4, RW);
        for (int t = 0; t < T; ++t)
            atg(a_p2e + (size_t)t * CP * CE, CE, CE, CP, ms_op, cat_o + 3 * H4 + t * H4, RW);
        atg(i2o_ent, CE, CE, NE, ms_ie, cat_o + 6 * H4, RW);
        mlp2(cat_o, CE, RW, RW, H, M_RECV_OE, hid_o, mr_oe, H, true);

        // ---- recv ont pred ----
        for (int t = 0; t < T; ++t)
            atg(a_e2p + (size_t)t * CE * CP, CP, CP, CE, ms_oe, cat_o + t * H4, RW);
        for (int t = 0; t < T; ++t)
            atg(a_p2p + (size_t)t * CP * CP, CP, CP, CP, ms_op, cat_o + 3 * H4 + t * H4, RW);
        atg(i2o_pred, CP, CP, NPRED, ms_ip, cat_o + 6 * H4, RW);
        mlp2(cat_o, CP, RW, RW, H, M_RECV_OP, hid_o, mr_op, H, true);

        // ---- recv img ent: [segsum(ms_ip,subj) | segsum(ms_ip,obj) | i2o_ent@ms_oe]
        zero(SB, (size_t)NE * 768);
        tg_scatter_pairs<<<dim3(NPRED), dim3(256), 0, stream>>>(pairs, ms_ip, SB, 768);
        gemm(i2o_ent, CE, 0, NE, CE, ms_oe, H4, 0, H4, nullptr, nullptr, 0, SB + 512, 768, 0);
        mlp2(SB, NE, 768, 768, H, M_RECV_IE, SA, SC, H, true);

        // ---- recv img pred: [ms_ie[subj] | ms_ie[obj] | i2o_pred@ms_op] ----
        tg_gather_pairs<<<dim3(NPRED), dim3(256), 0, stream>>>(pairs, ms_ie, SB, 768);
        gemm(i2o_pred, CP, 0, NPRED, CP, ms_op, H4, 0, H4, nullptr, nullptr, 0, SB + 512, 768, 0);
        mlp2(SB, NPRED, 768, 768, H, M_RECV_IP, SA, SD, H, true);

        // ---- GRU updates (SA/SB/SE as temps; SC=mr_ie, SD=mr_ip persist) ----
        gru(GRU_ONT_ENT,  ont_ent,  mr_oe, CE,    SA, SB, SE);
        gru(GRU_ONT_PRED, ont_pred, mr_op, CP,    SA, SB, SE);
        gru(GRU_IMG_ENT,  img_ent,  SC,    NE,    SA, SB, SE);
        gru(GRU_IMG_PRED, img_pred, SD,    NPRED, SA, SB, SE);

        // ---- outputs: rel_logits = mlp2(img_pred) @ mlp2(ont_pred,no-relu)^T ----
        mlp2(img_pred, NPRED, H, H, H, M_OUT_IP, SA, SB, H, true);
        mlp2(ont_pred, CP,    H, H, H, M_OUT_OP, cat_o, hid_o, H, false);
        gemm(SB, H, 0, NPRED, H, hid_o, H, 1, CP, nullptr, nullptr, 0, out, CP, 0);
        tg_softmax_rows<<<dim3(NPRED), dim3(128), 0, stream>>>(out, i2o_pred, CP);
    }
}
